// LSTM_76965813944958
// MI455X (gfx1250) — matrix operand order repaired and checked by reading, not compiled, not run on hardware
//
#include <hip/hip_runtime.h>
#include <hip/hip_bf16.h>
#include <stdint.h>

// ---------------------------------------------------------------------------
// 2-layer LSTM (B=128, T=256, IN=128, H=1024) + dense head, bf16 WMMA.
// Persistent kernel. Each wave32 owns one (row-tile, hidden-tile) pair and
// computes all 4 gate tiles (i,f,g,o) -> cell update is lane-local, cell
// state stays in registers for all 256 steps. Ping-pong h buffers give
// 2 grid barriers per timestep (down from 4).
// ---------------------------------------------------------------------------

typedef __attribute__((ext_vector_type(16))) __bf16 v16bf;
typedef __attribute__((ext_vector_type(8)))  float  v8f;
typedef __attribute__((ext_vector_type(4)))  unsigned int v4u;

constexpr int B_  = 128;
constexpr int T_  = 256;
constexpr int IN_ = 128;
constexpr int H_  = 1024;
constexpr int G_  = 4096;                 // 4*H (i,f,g,o)
constexpr int K0_ = IN_ + H_;             // 1152 (layer0 concat K)
constexpr int K1_ = H_ + H_;              // 2048 (layer1 concat K)

// main persistent grid: 64 blocks x 8 waves = 512 waves = 8 row-tiles x 64 hidden-tiles
constexpr int NBLK = 64;
constexpr int NTHR = 256;
constexpr int NTHREADS = NBLK * NTHR;     // 16384

// prep grid (bandwidth-bound, independent of persistent grid)
constexpr int PBLK = 256;
constexpr int PTHREADS = PBLK * NTHR;     // 65536

// ---- workspace layout (bytes) ----
constexpr size_t OFF_SYNC = 0;                                       // 2 u32 (count, gen)
constexpr size_t OFF_B0   = 256;                                     // fused bias l0, f32[4096]
constexpr size_t OFF_B1   = OFF_B0  + (size_t)G_ * 4;
constexpr size_t OFF_W0   = OFF_B1  + (size_t)G_ * 4;                // bf16 [4096][1152] (Wih0|Whh0)
constexpr size_t OFF_W1   = OFF_W0  + (size_t)G_ * K0_ * 2;          // bf16 [4096][2048] (Wih1|Whh1)
constexpr size_t OFF_XB   = OFF_W1  + (size_t)G_ * K1_ * 2;          // bf16 [T][B][IN] time-major
constexpr size_t OFF_H0B  = OFF_XB  + (size_t)T_ * B_ * IN_ * 2;     // bf16 [2][B][H] ping-pong
constexpr size_t OFF_H1B  = OFF_H0B + (size_t)2 * B_ * H_ * 2;       // bf16 [2][B][H] ping-pong
constexpr size_t OFF_H1F  = OFF_H1B + (size_t)2 * B_ * H_ * 2;       // f32  [B][H] final h1
constexpr size_t WS_TOTAL = OFF_H1F + (size_t)B_ * H_ * 4;           // ~36.2 MB

__device__ __forceinline__ unsigned short f2bf(float x) {
    unsigned u = __float_as_uint(x);
    unsigned r = u + 0x7FFFu + ((u >> 16) & 1u);   // round-to-nearest-even
    return (unsigned short)(r >> 16);
}

__device__ __forceinline__ float sigmoidf(float x) {
    return 1.0f / (1.0f + __expf(-x));
}

// Grid-wide barrier (monotonic generation counter).
__device__ __forceinline__ void grid_barrier(unsigned* cnt, unsigned* gen, unsigned* mygen) {
    __threadfence();                 // release: make this thread's stores visible (agent)
    __syncthreads();
    if (threadIdx.x == 0) {
        unsigned target = *mygen + 1;
        unsigned arrived = __hip_atomic_fetch_add(cnt, 1u, __ATOMIC_ACQ_REL,
                                                  __HIP_MEMORY_SCOPE_AGENT);
        if (arrived == (unsigned)(NBLK - 1)) {
            __hip_atomic_store(cnt, 0u, __ATOMIC_RELAXED, __HIP_MEMORY_SCOPE_AGENT);
            __hip_atomic_fetch_add(gen, 1u, __ATOMIC_RELEASE, __HIP_MEMORY_SCOPE_AGENT);
        } else {
            while (__hip_atomic_load(gen, __ATOMIC_ACQUIRE, __HIP_MEMORY_SCOPE_AGENT) < target) {
                __builtin_amdgcn_s_sleep(2);
            }
        }
        *mygen = target;
    }
    __syncthreads();
    __threadfence();                 // acquire: invalidate per-CU L0
}

// 4 gate tiles sharing one A stream: acc[g] += A[16x32k] * B_g[32k x 16].
// A rows split across two buffers: k < ksplit -> arow_a[k], else arow_b[k-ksplit].
// Fragment layouts per CDNA5 ISA 7.12.2 (wave32, 16-bit A 16x32 / B 32x16):
//   A: m = lane&15, h = lane>>4; V0-3 = k: kb+8h+0..7, V4-7 = k: kb+16+8h+0..7
//   B: n = lane&15;              V0-3 = k: kb+8h+0..7, V4-7 = k: kb+16+8h+0..7
__device__ __forceinline__ void wave_gemm4(const unsigned short* arow_a,
                                           const unsigned short* arow_b,
                                           int ksplit, int nk,
                                           const unsigned short* const wr[4],
                                           int half, v8f acc[4]) {
    for (int kc = 0; kc < nk; ++kc) {
        const int kb = kc * 32;
        const unsigned short* as = (kb < ksplit) ? (arow_a + kb) : (arow_b + (kb - ksplit));
        union { v4u u[2]; v16bf bf; } A;
        A.u[0] = *(const v4u*)(as + half * 8);          // V0-3
        A.u[1] = *(const v4u*)(as + 16 + half * 8);     // V4-7
        #pragma unroll
        for (int g = 0; g < 4; ++g) {
            union { v4u u[2]; v16bf bf; } Bf;
            const unsigned short* bs = wr[g] + kb + half * 8;
            Bf.u[0] = *(const v4u*)(bs);                // V0-3
            Bf.u[1] = *(const v4u*)(bs + 16);           // V4-7
            if (kc + 1 < nk) __builtin_prefetch(bs + 32, 0, 1);
            acc[g] = __builtin_amdgcn_wmma_f32_16x16x32_bf16(false, A.bf, false, Bf.bf,
                                                             (short)0, acc[g], false, false);
        }
    }
}

// ---------------------------------------------------------------------------
// Prep: zero sync/state, fuse biases, pack weights to bf16 concat layout,
// transpose x to time-major bf16.
// ---------------------------------------------------------------------------
__global__ void __launch_bounds__(NTHR)
lstm_prep(char* ws, const float* x,
          const float* Wih0, const float* Whh0, const float* bih0, const float* bhh0,
          const float* Wih1, const float* Whh1, const float* bih1, const float* bhh1) {
    const int gtid = blockIdx.x * NTHR + threadIdx.x;
    if (gtid == 0) { ((unsigned*)ws)[0] = 0u; ((unsigned*)ws)[1] = 0u; }

    float* b0 = (float*)(ws + OFF_B0);
    float* b1 = (float*)(ws + OFF_B1);
    for (int i = gtid; i < G_; i += PTHREADS) {
        b0[i] = bih0[i] + bhh0[i];
        b1[i] = bih1[i] + bhh1[i];
    }

    unsigned short* w0 = (unsigned short*)(ws + OFF_W0);
    for (int i = gtid; i < G_ * K0_; i += PTHREADS) {
        int r = i / K0_, c = i - r * K0_;
        float v = (c < IN_) ? Wih0[r * IN_ + c] : Whh0[r * H_ + (c - IN_)];
        w0[i] = f2bf(v);
    }
    unsigned short* w1 = (unsigned short*)(ws + OFF_W1);
    for (int i = gtid; i < G_ * K1_; i += PTHREADS) {
        int r = i / K1_, c = i - r * K1_;
        float v = (c < H_) ? Wih1[r * H_ + c] : Whh1[r * H_ + (c - H_)];
        w1[i] = f2bf(v);
    }

    unsigned short* xb = (unsigned short*)(ws + OFF_XB);
    for (int i = gtid; i < T_ * B_ * IN_; i += PTHREADS) {
        int t = i / (B_ * IN_);
        int rem = i - t * (B_ * IN_);
        int m = rem / IN_, k = rem - m * IN_;
        xb[i] = f2bf(x[((size_t)m * T_ + t) * IN_ + k]);   // x is [B][T][IN]
    }

    unsigned short* h0b = (unsigned short*)(ws + OFF_H0B);
    unsigned short* h1b = (unsigned short*)(ws + OFF_H1B);
    float* h1f = (float*)(ws + OFF_H1F);
    for (int i = gtid; i < 2 * B_ * H_; i += PTHREADS) {
        h0b[i] = 0; h1b[i] = 0;                  // ping buffer (index 0) must be zero at t=0
        if (i < B_ * H_) h1f[i] = 0.0f;
    }
}

// ---------------------------------------------------------------------------
// Persistent timestep loop.
//   block = hidden tile hct (0..63), wave-in-block = row tile rt (0..7).
//   Each wave: 4 gate accumulators + in-register cell state for its
//   16(batch) x 16(hidden) patch. 2 grid barriers per timestep.
// ---------------------------------------------------------------------------
__global__ void __launch_bounds__(NTHR)
lstm_main(char* ws, const float* Wd, const float* bd, float* out) {
    unsigned* cnt = (unsigned*)(ws + OFF_SYNC);
    unsigned* gen = cnt + 1;
    const float* b0 = (const float*)(ws + OFF_B0);
    const float* b1 = (const float*)(ws + OFF_B1);
    const unsigned short* w0 = (const unsigned short*)(ws + OFF_W0);
    const unsigned short* w1 = (const unsigned short*)(ws + OFF_W1);
    const unsigned short* xb = (const unsigned short*)(ws + OFF_XB);
    unsigned short* h0b = (unsigned short*)(ws + OFF_H0B);   // [2][B][H]
    unsigned short* h1b = (unsigned short*)(ws + OFF_H1B);   // [2][B][H]
    float* h1f = (float*)(ws + OFF_H1F);

    const int lane = threadIdx.x & 31;
    const int rt   = threadIdx.x >> 5;        // row tile 0..7
    const int hct  = blockIdx.x;              // hidden tile 0..63
    const int nrow = lane & 15;
    const int half = lane >> 4;
    const int gtid = blockIdx.x * NTHR + threadIdx.x;

    const int m    = rt * 16 + nrow;          // A-matrix row (batch index) for this lane
    const int jcol = hct * 16 + nrow;         // hidden column for this lane

    // per-gate weight rows + biases (gate g column = g*H + jcol)
    const unsigned short* w0r[4];
    const unsigned short* w1r[4];
    float bia0[4], bia1[4];
    #pragma unroll
    for (int g = 0; g < 4; ++g) {
        int col = g * H_ + jcol;
        w0r[g] = w0 + (size_t)col * K0_;
        w1r[g] = w1 + (size_t)col * K1_;
        bia0[g] = b0[col];
        bia1[g] = b1[col];
    }

    // in-register cell state for this wave's 16x16 patch (C/D layout)
    float c0reg[8], c1reg[8];
    #pragma unroll
    for (int v = 0; v < 8; ++v) { c0reg[v] = 0.0f; c1reg[v] = 0.0f; }

    unsigned mygen = 0;

    for (int t = 0; t < T_; ++t) {
        const int p = t & 1;
        const unsigned short* h0_rd = h0b + (size_t)p * B_ * H_;        // h0_{t-1}
        unsigned short*       h0_wr = h0b + (size_t)(p ^ 1) * B_ * H_;  // h0_t
        const unsigned short* h1_rd = h1b + (size_t)p * B_ * H_;        // h1_{t-1}
        unsigned short*       h1_wr = h1b + (size_t)(p ^ 1) * B_ * H_;  // h1_t

        // ---- layer0: gates = [x_t | h0_old] @ W0^T ; fused cell update ----
        {
            const unsigned short* xrow  = xb + ((size_t)t * B_ + m) * IN_;
            const unsigned short* h0row = h0_rd + (size_t)m * H_;
            v8f acc[4] = {{0.f,0.f,0.f,0.f,0.f,0.f,0.f,0.f},
                          {0.f,0.f,0.f,0.f,0.f,0.f,0.f,0.f},
                          {0.f,0.f,0.f,0.f,0.f,0.f,0.f,0.f},
                          {0.f,0.f,0.f,0.f,0.f,0.f,0.f,0.f}};
            wave_gemm4(xrow, h0row, IN_, K0_ / 32, w0r, half, acc);
            #pragma unroll
            for (int v = 0; v < 8; ++v) {
                float iv = sigmoidf(acc[0][v] + bia0[0]);
                float fv = sigmoidf(acc[1][v] + bia0[1]);
                float gv = tanhf   (acc[2][v] + bia0[2]);
                float ov = sigmoidf(acc[3][v] + bia0[3]);
                float c  = fv * c0reg[v] + iv * gv;
                c0reg[v] = c;
                float h  = ov * tanhf(c);
                int mm = rt * 16 + v + 8 * half;              // C/D: M = v + 8*(lane>>4)
                h0_wr[(size_t)mm * H_ + jcol] = f2bf(h);
            }
        }
        grid_barrier(cnt, gen, &mygen);

        // ---- layer1: gates = [h0_new | h1_old] @ W1^T ; fused cell update ----
        {
            const unsigned short* h0row = h0_wr + (size_t)m * H_;
            const unsigned short* h1row = h1_rd + (size_t)m * H_;
            v8f acc[4] = {{0.f,0.f,0.f,0.f,0.f,0.f,0.f,0.f},
                          {0.f,0.f,0.f,0.f,0.f,0.f,0.f,0.f},
                          {0.f,0.f,0.f,0.f,0.f,0.f,0.f,0.f},
                          {0.f,0.f,0.f,0.f,0.f,0.f,0.f,0.f}};
            wave_gemm4(h0row, h1row, H_, K1_ / 32, w1r, half, acc);
            #pragma unroll
            for (int v = 0; v < 8; ++v) {
                float iv = sigmoidf(acc[0][v] + bia1[0]);
                float fv = sigmoidf(acc[1][v] + bia1[1]);
                float gv = tanhf   (acc[2][v] + bia1[2]);
                float ov = sigmoidf(acc[3][v] + bia1[3]);
                float c  = fv * c1reg[v] + iv * gv;
                c1reg[v] = c;
                float h  = ov * tanhf(c);
                int mm = rt * 16 + v + 8 * half;
                h1_wr[(size_t)mm * H_ + jcol] = f2bf(h);
                if (t == T_ - 1) h1f[(size_t)mm * H_ + jcol] = h;   // fp32 for head
            }
        }
        grid_barrier(cnt, gen, &mygen);
    }

    // ---- dense head: out = tanh(h1 @ Wd^T + bd), [128 x 2] ----
    if (gtid < B_ * 2) {
        int mm = gtid >> 1, oo = gtid & 1;
        const float* hr = h1f + (size_t)mm * H_;
        const float* wr = Wd + (size_t)oo * H_;
        float s = bd[oo];
        for (int j = 0; j < H_; ++j) s += hr[j] * wr[j];
        out[gtid] = tanhf(s);
    }
}

extern "C" void kernel_launch(void* const* d_in, const int* in_sizes, int n_in,
                              void* d_out, int out_size, void* d_ws, size_t ws_size,
                              hipStream_t stream) {
    (void)in_sizes; (void)n_in; (void)out_size;
    if (ws_size < WS_TOTAL) return;   // workspace too small: cannot run safely

    const float* x    = (const float*)d_in[0];
    const float* Wih0 = (const float*)d_in[1];
    const float* Whh0 = (const float*)d_in[2];
    const float* bih0 = (const float*)d_in[3];
    const float* bhh0 = (const float*)d_in[4];
    const float* Wih1 = (const float*)d_in[5];
    const float* Whh1 = (const float*)d_in[6];
    const float* bih1 = (const float*)d_in[7];
    const float* bhh1 = (const float*)d_in[8];
    const float* Wd   = (const float*)d_in[9];
    const float* bd   = (const float*)d_in[10];

    char* ws = (char*)d_ws;
    lstm_prep<<<PBLK, NTHR, 0, stream>>>(ws, x, Wih0, Whh0, bih0, bhh0,
                                         Wih1, Whh1, bih1, bhh1);
    lstm_main<<<NBLK, NTHR, 0, stream>>>(ws, Wd, bd, (float*)d_out);
}
